// DiscreteMMSE_51178830299317
// MI455X (gfx1250) — matrix operand order repaired and checked by reading, not compiled, not run on hardware
//
#include <hip/hip_runtime.h>
#include <hip/hip_bf16.h>

typedef __attribute__((ext_vector_type(16))) __bf16 v16bf;
typedef __attribute__((ext_vector_type(8)))  float  v8f;
typedef __attribute__((ext_vector_type(4)))  float  f4v;

#define BATCH   64
#define NROW    256
#define DDIM    64
#define TTASK   4096
#define CHUNK   512
#define NCHUNK  8

// Build a v16bf WMMA operand from two contiguous 16B LDS reads (A/B layouts
// are arranged so each lane's 16 bf16 elements are two contiguous 8-element runs).
__device__ __forceinline__ v16bf ldAB(const __bf16* p, int e0, int e1) {
    union { f4v f[2]; v16bf v; } u;
    u.f[0] = *(const f4v*)(p + e0);
    u.f[1] = *(const f4v*)(p + e1);
    return u.v;
}

__global__ __launch_bounds__(256)
void mmse_flash_kernel(const float* __restrict__ data,
                       const float* __restrict__ targets,
                       const float* __restrict__ task_pool,
                       float* __restrict__ ws) {
    const int b    = blockIdx.x;
    const int c    = blockIdx.y;
    const int tid  = threadIdx.x;
    const int wave = tid >> 5;
    const int lane = tid & 31;
    const int half = lane >> 4;
    const int c16  = lane & 15;

    __shared__ __align__(16) __bf16 sW_rm[CHUNK * 64];   // [tloc][d]  (GEMM1 B: K=d contiguous)
    __shared__ __align__(16) __bf16 sW_dm[64 * CHUNK];   // [d][tloc]  (GEMM2 A: K=t contiguous)
    __shared__ __align__(16) __bf16 sData[NROW * 64];    // [n][d]     (GEMM1 A)
    __shared__ float  sTgt[NROW];
    __shared__ __align__(16) __bf16 sP[8][16 * 64];      // per-wave exp(P) staging [i][tloc]
    __shared__ float  sComb[8][16][66];                  // per-wave partials: acc[64], m, l
    __shared__ float  sScale[16][8];

    // ---- stage inputs into LDS (bf16) ----
    for (int k = tid; k < NROW * 64; k += 256)
        sData[k] = (__bf16)data[b * NROW * 64 + k];
    sTgt[tid] = targets[b * NROW + tid];
    for (int k = tid; k < CHUNK * 64; k += 256) {
        int tl = k >> 6, d = k & 63;
        float w = task_pool[(c * CHUNK + tl) * 64 + d];
        __bf16 wb = (__bf16)w;
        sW_rm[tl * 64 + d]    = wb;
        sW_dm[d * CHUNK + tl] = wb;
    }
    __syncthreads();

    const int wbase = wave * 64;           // this wave's 64 t-columns (chunk-local)
    float carry[4] = {0.f, 0.f, 0.f, 0.f}; // cumsum carry per 16-col tile (replicated in halves)

    for (int m = 0; m < 16; ++m) {         // n-tiles of 16 rows
        // GEMM1 A operands: rows m*16..m*16+15, K=d (0..63)
        const int arow = m * 16 + c16;
        v16bf A1 = ldAB(sData, arow * 64 +      half * 8, arow * 64 + 16 + half * 8);
        v16bf A2 = ldAB(sData, arow * 64 + 32 + half * 8, arow * 64 + 48 + half * 8);
        float tgt[8];
#pragma unroll
        for (int v = 0; v < 8; ++v) tgt[v] = sTgt[m * 16 + half * 8 + v];

        float s[4][8];
#pragma unroll
        for (int j = 0; j < 4; ++j) {
            const int tcol = wbase + j * 16 + c16;
            v16bf B1 = ldAB(sW_rm, tcol * 64 +      half * 8, tcol * 64 + 16 + half * 8);
            v16bf B2 = ldAB(sW_rm, tcol * 64 + 32 + half * 8, tcol * 64 + 48 + half * 8);
            v8f z = {0.f,0.f,0.f,0.f,0.f,0.f,0.f,0.f};
            z = __builtin_amdgcn_wmma_f32_16x16x32_bf16(false, A1, false, B1, (short)0, z, false, false);
            z = __builtin_amdgcn_wmma_f32_16x16x32_bf16(false, A2, false, B2, (short)0, z, false, false);

            // cumulative -0.5*r^2 down the 16 rows of this tile (C layout: 8 rows/lane-half)
            float p = 0.f;
#pragma unroll
            for (int v = 0; v < 8; ++v) {
                float r = tgt[v] - z[v];
                p = fmaf(-0.5f * r, r, p);
                s[j][v] = p;                       // prefix within this lane-half
            }
            float tot = __shfl_xor(p, 16);         // upper half gets lower half's rows 0..7 total
            float add = carry[j] + (half ? tot : 0.f);
#pragma unroll
            for (int v = 0; v < 8; ++v) s[j][v] += add;
            float top  = s[j][7];                  // upper lanes: cum through row 15
            float topx = __shfl_xor(top, 16);
            carry[j] = half ? top : topx;          // both halves now hold per-column carry
        }

        // per-row max over this wave's 64 columns (replicated across each 16-lane half)
        float mv[8], lv[8];
#pragma unroll
        for (int v = 0; v < 8; ++v) {
            float mx = s[0][v];
#pragma unroll
            for (int j = 1; j < 4; ++j) mx = fmaxf(mx, s[j][v]);
#pragma unroll
            for (int off = 1; off < 16; off <<= 1) mx = fmaxf(mx, __shfl_xor(mx, off));
            mv[v] = mx;
        }
        // exp + per-row sum
#pragma unroll
        for (int v = 0; v < 8; ++v) {
            float ls = 0.f;
#pragma unroll
            for (int j = 0; j < 4; ++j) { float e = __expf(s[j][v] - mv[v]); s[j][v] = e; ls += e; }
#pragma unroll
            for (int off = 1; off < 16; off <<= 1) ls += __shfl_xor(ls, off);
            lv[v] = ls;
        }

        // stage P (bf16, row-major [i][tloc]) for GEMM2 B operand
#pragma unroll
        for (int j = 0; j < 4; ++j)
#pragma unroll
            for (int v = 0; v < 8; ++v)
                sP[wave][(half * 8 + v) * 64 + j * 16 + c16] = (__bf16)s[j][v];
        asm volatile("s_wait_dscnt 0" ::: "memory");   // wave-private LDS RAW across lanes

        // GEMM2: accT(d,i) = W(d x t) * P^T(t x i), K = 64 columns of this wave
#pragma unroll
        for (int dt = 0; dt < 4; ++dt) {
            v8f acc = {0.f,0.f,0.f,0.f,0.f,0.f,0.f,0.f};
            const int drow = dt * 16 + c16;
#pragma unroll
            for (int kk = 0; kk < 2; ++kk) {
                const int tb = wbase + kk * 32;
                v16bf Aw = ldAB(sW_dm, drow * CHUNK + tb + half * 8,
                                       drow * CHUNK + tb + 16 + half * 8);
                v16bf Bp = ldAB(sP[wave], c16 * 64 + kk * 32 + half * 8,
                                          c16 * 64 + kk * 32 + 16 + half * 8);
                acc = __builtin_amdgcn_wmma_f32_16x16x32_bf16(false, Aw, false, Bp, (short)0, acc, false, false);
            }
#pragma unroll
            for (int v = 0; v < 8; ++v)
                sComb[wave][c16][dt * 16 + half * 8 + v] = acc[v];   // D: N=i=c16, M=d
        }
        if (c16 == 0) {
#pragma unroll
            for (int v = 0; v < 8; ++v) {
                sComb[wave][half * 8 + v][64] = mv[v];
                sComb[wave][half * 8 + v][65] = lv[v];
            }
        }
        __syncthreads();

        // combine the 8 wave partials -> per-(b,chunk) partial in ws
        const size_t wsbase = (((size_t)b * NCHUNK + c) * NROW + m * 16) * 66;
        if (tid < 16) {
            const int row = tid;
            float M = sComb[0][row][64];
#pragma unroll
            for (int w = 1; w < 8; ++w) M = fmaxf(M, sComb[w][row][64]);
            float L = 0.f;
#pragma unroll
            for (int w = 0; w < 8; ++w) {
                float sc = __expf(sComb[w][row][64] - M);
                sScale[row][w] = sc;
                L += sc * sComb[w][row][65];
            }
            ws[wsbase + row * 66 + 64] = M;
            ws[wsbase + row * 66 + 65] = L;
        }
        __syncthreads();
        for (int k = tid; k < 16 * 64; k += 256) {
            const int row = k >> 6, d = k & 63;
            float a = 0.f;
#pragma unroll
            for (int w = 0; w < 8; ++w) a += sScale[row][w] * sComb[w][row][d];
            ws[wsbase + row * 66 + d] = a;
        }
        __syncthreads();
    }
}

__global__ __launch_bounds__(256)
void mmse_finalize_kernel(const float* __restrict__ data,
                          const float* __restrict__ task_pool,
                          const float* __restrict__ ws,
                          float* __restrict__ out) {
    const int b = blockIdx.x, tid = threadIdx.x;
    __shared__ float sPart[4][64];
    __shared__ float sWm[64];

    // W.mean(axis=t), cooperatively
    const int d = tid & 63, g = tid >> 6;
    float sum = 0.f;
    for (int t = g; t < TTASK; t += 4) sum += task_pool[t * 64 + d];
    sPart[g][d] = sum;
    __syncthreads();
    if (tid < 64)
        sWm[tid] = (sPart[0][tid] + sPart[1][tid] + sPart[2][tid] + sPart[3][tid]) * (1.0f / TTASK);
    __syncthreads();

    if (tid == 0) {
        float p = 0.f;
        for (int dd = 0; dd < 64; ++dd) p += data[(b * NROW) * 64 + dd] * sWm[dd];
        out[b * NROW] = p;
    } else {
        const int i = tid - 1;                       // alpha row 0..254
        const size_t base = ((size_t)b * NCHUNK * NROW + i) * 66;
        const size_t cs   = (size_t)NROW * 66;
        float M = ws[base + 64];
#pragma unroll
        for (int cc = 1; cc < NCHUNK; ++cc) M = fmaxf(M, ws[base + cc * cs + 64]);
        float sc[NCHUNK], L = 0.f;
#pragma unroll
        for (int cc = 0; cc < NCHUNK; ++cc) {
            sc[cc] = __expf(ws[base + cc * cs + 64] - M);
            L += sc[cc] * ws[base + cc * cs + 65];
        }
        const float inv = 1.0f / L;
        float pred = 0.f;
        for (int dd = 0; dd < 64; ++dd) {
            float a = 0.f;
#pragma unroll
            for (int cc = 0; cc < NCHUNK; ++cc) a += sc[cc] * ws[base + cc * cs + dd];
            pred += data[(b * NROW + tid) * 64 + dd] * a;   // data row i+1 == tid
        }
        out[b * NROW + tid] = pred * inv;
    }
}

extern "C" void kernel_launch(void* const* d_in, const int* in_sizes, int n_in,
                              void* d_out, int out_size, void* d_ws, size_t ws_size,
                              hipStream_t stream) {
    (void)in_sizes; (void)n_in; (void)out_size; (void)ws_size;
    const float* data      = (const float*)d_in[0];
    const float* targets   = (const float*)d_in[1];
    const float* task_pool = (const float*)d_in[2];
    float* out = (float*)d_out;
    float* ws  = (float*)d_ws;   // needs 64*8*256*66*4 = 34,603,008 bytes

    dim3 g1(BATCH, NCHUNK);
    mmse_flash_kernel<<<g1, 256, 0, stream>>>(data, targets, task_pool, ws);
    mmse_finalize_kernel<<<BATCH, 256, 0, stream>>>(data, task_pool, ws, out);
}
